// DistributedGIN_30520037606035
// MI455X (gfx1250) — matrix operand order repaired and checked by reading, not compiled, not run on hardware
//
#include <hip/hip_runtime.h>

// Problem constants (from reference)
#define NN   50000
#define NE   800000
#define CIN  128
#define CHID 256
#define COUT 64
#define BN_EPS 1e-5f

typedef __attribute__((ext_vector_type(16))) __bf16 v16bf;
typedef __attribute__((ext_vector_type(8)))  float  v8f;

// ---------------------------------------------------------------------------
// Elementwise helpers
// ---------------------------------------------------------------------------
__global__ void k_copy4(const float4* __restrict__ a, float4* __restrict__ b, int n4) {
    int i = blockIdx.x * blockDim.x + threadIdx.x;
    if (i < n4) b[i] = a[i];
}

__global__ void k_zero(float* __restrict__ p, int n) {
    int i = blockIdx.x * blockDim.x + threadIdx.x;
    if (i < n) p[i] = 0.f;
}

// Transpose + convert weight W[K x NCc] (f32 row-major) -> Wt[NCc x K] (bf16)
__global__ void k_wt_bf16(const float* __restrict__ W, __bf16* __restrict__ Wt,
                          int K, int NCc) {
    int i = blockIdx.x * blockDim.x + threadIdx.x;
    if (i >= K * NCc) return;
    int k = i / NCc;
    int n = i - k * NCc;
    Wt[(size_t)n * K + k] = (__bf16)W[i];
}

// ---------------------------------------------------------------------------
// CSR build (once per launch; edge_index is layer-invariant)
// ---------------------------------------------------------------------------
__global__ void k_count(const int* __restrict__ dst, int* __restrict__ deg) {
    int e = blockIdx.x * blockDim.x + threadIdx.x;
    if (e < NE) atomicAdd(&deg[dst[e]], 1);
}

// Exclusive scan of deg[0..NN) -> rowptr[0..NN]; single 1024-thread block.
__global__ __launch_bounds__(1024) void k_scan(const int* __restrict__ deg,
                                               int* __restrict__ rowptr) {
    __shared__ int sums[1024];
    const int t = threadIdx.x;
    const int chunk = (NN + 1023) / 1024;            // 49
    int lo = t * chunk;
    int hiE = lo + chunk; if (hiE > NN) hiE = NN;
    int s = 0;
    for (int i = lo; i < hiE; ++i) s += deg[i];
    sums[t] = s;
    __syncthreads();
    if (t == 0) {                                    // serial scan of 1024 partials
        int run = 0;
        for (int i = 0; i < 1024; ++i) { int v = sums[i]; sums[i] = run; run += v; }
        rowptr[NN] = run;
    }
    __syncthreads();
    int run = sums[t];
    for (int i = lo; i < hiE; ++i) { int v = deg[i]; rowptr[i] = run; run += v; }
}

__global__ void k_fill(const int* __restrict__ src, const int* __restrict__ dst,
                       int* __restrict__ cursor, int* __restrict__ colidx) {
    int e = blockIdx.x * blockDim.x + threadIdx.x;
    if (e >= NE) return;
    int pos = atomicAdd(&cursor[dst[e]], 1);
    colidx[pos] = src[e];
}

// ---------------------------------------------------------------------------
// GIN aggregation by gather: y[n] = x[n] + sum_{j in nbrs(n)} x[j]   (eps = 0)
// One wave per node; lanes own float4 channel chunks (register accumulate).
// ---------------------------------------------------------------------------
template <int DIM4>   // float4 chunks per row: 32 (CIN) or 64 (CHID)
__global__ __launch_bounds__(256) void k_gather_agg(const float* __restrict__ x,
        const int* __restrict__ rowptr, const int* __restrict__ colidx,
        float* __restrict__ y) {
    constexpr int PER = DIM4 / 32;                   // chunks per lane (1 or 2)
    const int wv   = threadIdx.x >> 5;
    const int lane = threadIdx.x & 31;
    const int node = blockIdx.x * 8 + wv;            // NN % 8 == 0
    if (node >= NN) return;

    const float4* xr = (const float4*)x + (size_t)node * DIM4;
    float4 acc[PER];
#pragma unroll
    for (int p = 0; p < PER; ++p) acc[p] = xr[lane + 32 * p];

    const int beg = rowptr[node], end = rowptr[node + 1];
    for (int j = beg; j < end; ++j) {
        int nb = colidx[j];
        const float4* nr = (const float4*)x + (size_t)nb * DIM4;
#pragma unroll
        for (int p = 0; p < PER; ++p) {
            float4 v = nr[lane + 32 * p];
            acc[p].x += v.x; acc[p].y += v.y; acc[p].z += v.z; acc[p].w += v.w;
        }
    }
    float4* yr = (float4*)y + (size_t)node * DIM4;
#pragma unroll
    for (int p = 0; p < PER; ++p) yr[lane + 32 * p] = acc[p];
}

// ---------------------------------------------------------------------------
// BatchNorm: two-pass (partial sums via atomics -> finalize -> apply)
// ---------------------------------------------------------------------------
__global__ void k_bn_stats(const float* __restrict__ h, float* __restrict__ ssum,
                           float* __restrict__ ssq) {
    int c = threadIdx.x;                             // blockDim == CHID
    float s = 0.f, q = 0.f;
    for (int r = blockIdx.x; r < NN; r += gridDim.x) {
        float v = h[(size_t)r * CHID + c];
        s += v;
        q += v * v;
    }
    atomicAdd(&ssum[c], s);
    atomicAdd(&ssq[c], q);
}

__global__ void k_bn_finalize(const float* __restrict__ ssum, const float* __restrict__ ssq,
                              const float* __restrict__ gamma, const float* __restrict__ beta,
                              float* __restrict__ scale, float* __restrict__ shift) {
    int c = threadIdx.x;
    float inv_n = 1.f / (float)NN;
    float mean = ssum[c] * inv_n;
    float var  = ssq[c] * inv_n - mean * mean;       // population variance (ddof=0)
    float sc   = gamma[c] * rsqrtf(var + BN_EPS);
    scale[c] = sc;
    shift[c] = beta[c] - mean * sc;
}

__global__ void k_bn_apply_relu4(float4* __restrict__ h, const float4* __restrict__ scale,
                                 const float4* __restrict__ shift, int n4) {
    int i = blockIdx.x * blockDim.x + threadIdx.x;
    if (i >= n4) return;
    int c4 = i & (CHID / 4 - 1);
    float4 v = h[i];
    float4 sc = scale[c4];
    float4 sh = shift[c4];
    v.x = fmaxf(fmaf(v.x, sc.x, sh.x), 0.f);
    v.y = fmaxf(fmaf(v.y, sc.y, sh.y), 0.f);
    v.z = fmaxf(fmaf(v.z, sc.z, sh.z), 0.f);
    v.w = fmaxf(fmaf(v.w, sc.w, sh.w), 0.f);
    h[i] = v;
}

// ---------------------------------------------------------------------------
// Fully-unrolled software-pipelined WMMA GEMM (K compile-time: 128 or 256):
// C[M x NC_] = A[M x K] @ W[K x NC_] + bias (+ relu).
//  * Weights pre-converted to bf16, transposed: Wt[n][k].
//  * LDS weight tile double-buffered; one barrier per K-step.
//  * All NT B-fragments burst-loaded before the WMMA chain (partial dscnt waits).
//  * Full unroll removes register-rotation movs and constant-folds addresses.
// ---------------------------------------------------------------------------
template <int NC_, int CSPLIT, int K, int RELU>
__global__ __launch_bounds__(128 * CSPLIT) void k_gemm_wmma(
    const float* __restrict__ A, const __bf16* __restrict__ Wt,
    const float* __restrict__ bias, float* __restrict__ C, int M) {
    constexpr int NT      = (NC_ / 16) / CSPLIT;     // column tiles per wave
    constexpr int THREADS = 128 * CSPLIT;
    constexpr int SEGS    = (NC_ * 4) / THREADS;     // 16B segments per thread
    constexpr int KSTEPS  = K / 32;
    __shared__ __attribute__((aligned(32))) __bf16 Bs[2][NC_ * 32];  // [n][k]

    const int tid  = threadIdx.x;
    const int lane = tid & 31;
    const int w    = tid >> 5;
    const int strip_local = w / CSPLIT;
    const int colhalf     = w % CSPLIT;

    const int strips = M >> 4;                       // M % 16 == 0
    int strip = blockIdx.x * 4 + strip_local;
    const bool valid = (strip < strips);
    if (!valid) strip = strips - 1;                  // clamp loads; guard stores

    const int hi      = lane >> 4;
    const int rloc    = lane & 15;
    const int kbaseA  = hi * 16;                     // A frag K: 0-15 | 16-31
    const int kbaseB  = hi * 16;                     // B frag K: 0-15 | 16-31
    const int colbase = colhalf * (NT * 16);
    const float* abase = A + (size_t)(strip * 16 + rloc) * K;

    v8f zero = {};
    v8f acc[NT];
#pragma unroll
    for (int t = 0; t < NT; ++t) acc[t] = zero;

    // ---- prologue: stage k-step 0 into Bs[0]; prefetch A(k = 0)
#pragma unroll
    for (int j = 0; j < SEGS; ++j) {
        int s = tid + THREADS * j;
        int n = s >> 2, part = s & 3;
        uint4 v = ((const uint4*)(Wt + (size_t)n * K))[part];
        ((uint4*)&Bs[0][n * 32])[part] = v;
    }
    float4 a0 = ((const float4*)(abase + kbaseA))[0];
    float4 a1 = ((const float4*)(abase + kbaseA))[1];
    float4 a2 = ((const float4*)(abase + kbaseA))[2];
    float4 a3 = ((const float4*)(abase + kbaseA))[3];
    __syncthreads();

#pragma unroll
    for (int s = 0; s < KSTEPS; ++s) {
        const int cur   = s & 1;
        const int knext = ((s + 1 < KSTEPS) ? (s + 1) : (KSTEPS - 1)) * 32;

        // Prefetch next weight tile into registers
        uint4 g[SEGS];
#pragma unroll
        for (int j = 0; j < SEGS; ++j) {
            int si = tid + THREADS * j;
            int n = si >> 2, part = si & 3;
            g[j] = ((const uint4*)(Wt + (size_t)n * K + knext))[part];
        }
        // Prefetch next A fragment
        const float* ar = abase + knext;
        float4 n0 = ((const float4*)(ar + kbaseA))[0];
        float4 n1 = ((const float4*)(ar + kbaseA))[1];
        float4 n2 = ((const float4*)(ar + kbaseA))[2];
        float4 n3 = ((const float4*)(ar + kbaseA))[3];

        // Build current A fragment (packed f32 -> bf16 converts)
        v16bf af;
        af[0]  = (__bf16)a0.x; af[1]  = (__bf16)a0.y; af[2]  = (__bf16)a0.z; af[3]  = (__bf16)a0.w;
        af[4]  = (__bf16)a1.x; af[5]  = (__bf16)a1.y; af[6]  = (__bf16)a1.z; af[7]  = (__bf16)a1.w;
        af[8]  = (__bf16)a2.x; af[9]  = (__bf16)a2.y; af[10] = (__bf16)a2.z; af[11] = (__bf16)a2.w;
        af[12] = (__bf16)a3.x; af[13] = (__bf16)a3.y; af[14] = (__bf16)a3.z; af[15] = (__bf16)a3.w;

        // Burst-load all B fragments, then run the WMMA chain
        v16bf bfr[NT];
#pragma unroll
        for (int t = 0; t < NT; ++t)
            bfr[t] = *(const v16bf*)&Bs[cur][(colbase + t * 16 + rloc) * 32 + kbaseB];
#pragma unroll
        for (int t = 0; t < NT; ++t)
            acc[t] = __builtin_amdgcn_wmma_f32_16x16x32_bf16(
                false, af, false, bfr[t], (short)0, acc[t], false, false);

        // Dump staged weight tile into the other LDS buffer
#pragma unroll
        for (int j = 0; j < SEGS; ++j) {
            int si = tid + THREADS * j;
            int n = si >> 2, part = si & 3;
            ((uint4*)&Bs[cur ^ 1][n * 32])[part] = g[j];
        }
        __syncthreads();
        a0 = n0; a1 = n1; a2 = n2; a3 = n3;          // SSA renames (full unroll)
    }

    if (!valid) return;
    // C/D layout: VGPR j, lanes0-15 -> M=j, lanes16-31 -> M=8+j; col = lane%16
#pragma unroll
    for (int t = 0; t < NT; ++t) {
        int col = colbase + t * 16 + rloc;
        float b = bias[col];
#pragma unroll
        for (int j = 0; j < 8; ++j) {
            int row = strip * 16 + hi * 8 + j;
            float v = acc[t][j] + b;
            if (RELU) v = fmaxf(v, 0.f);
            C[(size_t)row * NC_ + col] = v;
        }
    }
}

// ---------------------------------------------------------------------------
// Host orchestration
// ---------------------------------------------------------------------------
#define GEMM_BLOCKS ((NN / 16 + 3) / 4)

extern "C" void kernel_launch(void* const* d_in, const int* in_sizes, int n_in,
                              void* d_out, int out_size, void* d_ws, size_t ws_size,
                              hipStream_t stream) {
    (void)in_sizes; (void)n_in; (void)out_size; (void)ws_size;

    // Inputs flattened in setup_inputs() insertion order
    const float* x   = (const float*)d_in[0];    // (N, 128)
    const int*   ei  = (const int*)d_in[1];      // (2, E) row-major
    const int*   src = ei;
    const int*   dst = ei + NE;

    struct Conv { const float *w1, *b1, *gamma, *beta, *w2, *b2; };
    Conv cv[3];
    int p = 2;
    for (int i = 0; i < 3; ++i) {
        cv[i].w1    = (const float*)d_in[p++];
        cv[i].b1    = (const float*)d_in[p++];
        cv[i].gamma = (const float*)d_in[p++];
        cv[i].beta  = (const float*)d_in[p++];
        cv[i].w2    = (const float*)d_in[p++];
        cv[i].b2    = (const float*)d_in[p++];
    }
    const float* cw1 = (const float*)d_in[p++];
    const float* cb1 = (const float*)d_in[p++];
    const float* cw2 = (const float*)d_in[p++];
    const float* cb2 = (const float*)d_in[p++];

    // Workspace layout (all regions 16B-aligned)
    const size_t BUF = (size_t)NN * CHID;
    float* bufA  = (float*)d_ws;
    float* bufB  = bufA + BUF;
    float* ssum  = bufB + BUF;                   // 256
    float* ssq   = ssum + CHID;
    float* scale = ssq  + CHID;
    float* shift = scale + CHID;
    __bf16* wt0  = (__bf16*)(shift + CHID);      // 442368 bf16 total (884736 B)
    int* deg     = (int*)(wt0 + 376832);         // NN
    int* rowptr  = deg + NN;                     // NN + 4 (padded)
    int* cursor  = rowptr + NN + 4;              // NN
    int* colidx  = cursor + NN;                  // NE

    // ---- bf16 transposed weights ----
    __bf16* wtp = wt0;
    __bf16* wt_w1[3]; __bf16* wt_w2[3]; __bf16* wt_c1; __bf16* wt_c2;
    auto conv_wt = [&](const float* W, int K, int NCc) {
        __bf16* dstp = wtp;
        int n = K * NCc;
        k_wt_bf16<<<(n + 255) / 256, 256, 0, stream>>>(W, dstp, K, NCc);
        wtp += n;
        return dstp;
    };
    wt_w1[0] = conv_wt(cv[0].w1, CIN,  CHID);
    wt_w2[0] = conv_wt(cv[0].w2, CHID, CHID);
    wt_w1[1] = conv_wt(cv[1].w1, CHID, CHID);
    wt_w2[1] = conv_wt(cv[1].w2, CHID, CHID);
    wt_w1[2] = conv_wt(cv[2].w1, CHID, CHID);
    wt_w2[2] = conv_wt(cv[2].w2, CHID, CHID);
    wt_c1    = conv_wt(cw1,      CHID, CHID);
    wt_c2    = conv_wt(cw2,      CHID, COUT);

    // ---- CSR build (reused by all layers) ----
    k_zero<<<(NN + 255) / 256, 256, 0, stream>>>((float*)deg, NN);
    k_count<<<(NE + 255) / 256, 256, 0, stream>>>(dst, deg);
    k_scan<<<1, 1024, 0, stream>>>(deg, rowptr);
    k_copy4<<<(NN / 4 + 255) / 256, 256, 0, stream>>>(
        (const float4*)rowptr, (float4*)cursor, NN / 4);
    k_fill<<<(NE + 255) / 256, 256, 0, stream>>>(src, dst, cursor, colidx);

    auto bn = [&](float* h, const float* gamma, const float* beta) {
        k_zero<<<1, 2 * CHID, 0, stream>>>(ssum, 2 * CHID);
        k_bn_stats<<<256, CHID, 0, stream>>>(h, ssum, ssq);
        k_bn_finalize<<<1, CHID, 0, stream>>>(ssum, ssq, gamma, beta, scale, shift);
        int n4 = NN * CHID / 4;
        k_bn_apply_relu4<<<(n4 + 255) / 256, 256, 0, stream>>>(
            (float4*)h, (const float4*)scale, (const float4*)shift, n4);
    };

    // ---- Layer 0 (din = 128): agg -> bufB, hidden -> bufA ----
    k_gather_agg<CIN / 4><<<NN / 8, 256, 0, stream>>>(x, rowptr, colidx, bufB);
    k_gemm_wmma<CHID, 2, CIN, 0><<<GEMM_BLOCKS, 256, 0, stream>>>(
        bufB, wt_w1[0], cv[0].b1, bufA, NN);
    bn(bufA, cv[0].gamma, cv[0].beta);
    k_gemm_wmma<CHID, 2, CHID, 1><<<GEMM_BLOCKS, 256, 0, stream>>>(
        bufA, wt_w2[0], cv[0].b2, bufB, NN);                 // + model-level relu

    // ---- Layers 1, 2 (din = 256), ping-pong bufB <-> bufA ----
    float* xin = bufB;
    float* tmp = bufA;
    for (int l = 1; l < 3; ++l) {
        k_gather_agg<CHID / 4><<<NN / 8, 256, 0, stream>>>(xin, rowptr, colidx, tmp);
        k_gemm_wmma<CHID, 2, CHID, 0><<<GEMM_BLOCKS, 256, 0, stream>>>(
            tmp, wt_w1[l], cv[l].b1, xin, NN);               // old x dead -> reuse
        bn(xin, cv[l].gamma, cv[l].beta);
        k_gemm_wmma<CHID, 2, CHID, 1><<<GEMM_BLOCKS, 256, 0, stream>>>(
            xin, wt_w2[l], cv[l].b2, tmp, NN);
        float* t = xin; xin = tmp; tmp = t;
    }

    // ---- Classifier head ----
    k_gemm_wmma<CHID, 2, CHID, 1><<<GEMM_BLOCKS, 256, 0, stream>>>(
        xin, wt_c1, cb1, tmp, NN);
    k_gemm_wmma<COUT, 1, CHID, 0><<<GEMM_BLOCKS, 128, 0, stream>>>(
        tmp, wt_c2, cb2, (float*)d_out, NN);
}
